// SVDLinear_3453153706201
// MI455X (gfx1250) — matrix operand order repaired and checked by reading, not compiled, not run on hardware
//
#include <hip/hip_runtime.h>

typedef __bf16 bf16_t;
typedef __attribute__((ext_vector_type(4)))  __bf16 v4bf;
typedef __attribute__((ext_vector_type(8)))  __bf16 v8bf;
typedef __attribute__((ext_vector_type(16))) __bf16 v16bf;
typedef __attribute__((ext_vector_type(4)))  float  v4f;
typedef __attribute__((ext_vector_type(8)))  float  v8f;

constexpr int TOKENS = 16384;
constexpr int D_IN   = 4096;
constexpr int D_OUT  = 4096;
constexpr int RANK   = 512;

// ---------------------------------------------------------------------------
// Prep kernel 1: Us[n][r] = bf16(U[n][r] * sigma[r])   (layout: [D_OUT][RANK])
// ---------------------------------------------------------------------------
__global__ __launch_bounds__(256)
void k_scale_u(const float* __restrict__ U, const float* __restrict__ sigma,
               bf16_t* __restrict__ Us) {
    int idx = blockIdx.x * 256 + threadIdx.x;          // 0 .. D_OUT*RANK-1
    int r   = idx & (RANK - 1);
    Us[idx] = (bf16_t)(U[idx] * sigma[r]);
}

// ---------------------------------------------------------------------------
// Prep kernel 2: Vt[r][k] = bf16(V[k][r])   V:[D_IN][RANK] -> Vt:[RANK][D_IN]
// LDS 32x32 tile transpose (padded to kill bank conflicts).
// ---------------------------------------------------------------------------
__global__ __launch_bounds__(256)
void k_transpose_v(const float* __restrict__ V, bf16_t* __restrict__ Vt) {
    __shared__ bf16_t tile[32][33];
    const int kb = blockIdx.x * 32;                    // along D_IN
    const int rb = blockIdx.y * 32;                    // along RANK
    const int tx = threadIdx.x & 31;
    const int ty = threadIdx.x >> 5;                   // 0..7
#pragma unroll
    for (int i = 0; i < 4; ++i) {
        int row = ty + 8 * i;                          // k within tile
        tile[row][tx] = (bf16_t)V[(size_t)(kb + row) * RANK + (rb + tx)];
    }
    __syncthreads();
#pragma unroll
    for (int i = 0; i < 4; ++i) {
        int row = ty + 8 * i;                          // r within tile
        Vt[(size_t)(rb + row) * D_IN + (kb + tx)] = tile[tx][row];
    }
}

// ---------------------------------------------------------------------------
// WMMA GEMM:  Out[M][N] = A[M][K] * Bt[N][K]^T
//   A_F32:    A is f32 (converted to bf16 on the way into LDS), else bf16.
//   OUT_BF16: result stored as bf16, else f32 (nontemporal stores).
// Tiles: BM=128, BN=128, BK=64; 256 threads = 8 waves (4 in M x 2 in N);
// each wave computes a 32x64 tile = 2x4 WMMA 16x16 accumulators.
// Double-buffered LDS: global loads for tile k+1 issued before the WMMAs of
// tile k, drained to the alternate buffer afterwards; 1 barrier per K-tile.
// ---------------------------------------------------------------------------
template <bool A_F32, bool OUT_BF16>
__global__ __launch_bounds__(256)
void k_gemm(const void* __restrict__ Aptr, const bf16_t* __restrict__ Bt,
            void* __restrict__ Outp, int M, int N, int K) {
    constexpr int BM = 128, BN = 128, BK = 64;
    constexpr int LDT = 72;                            // padded row (bf16 units)
    __shared__ __align__(16) bf16_t sA[2][BM * LDT];
    __shared__ __align__(16) bf16_t sB[2][BN * LDT];

    const int tid  = threadIdx.x;
    const int bm   = blockIdx.y * BM;
    const int bn   = blockIdx.x * BN;
    const int wave = tid >> 5;
    const int lane = tid & 31;
    const int wm   = (wave & 3) * 32;                  // wave M offset in block
    const int wn   = (wave >> 2) * 64;                 // wave N offset in block
    const int lm   = lane & 15;
    const int lh   = lane >> 4;                        // lane-half select

    // Per-thread staging registers for the in-flight (k+1) tile.
    v4f  a_stage_f[8];
    v8bf a_stage_h[4];
    v8bf b_stage[4];

    const int a_c4 = (tid & 15) * 4;                   // f32-A path mapping
    const int a_r0 = tid >> 4;
    const int c8   = (tid & 7) * 8;                    // bf16 path mapping
    const int r0   = tid >> 3;

    auto load_a = [&](int k0) {
        if constexpr (A_F32) {
            const float* A = (const float*)Aptr;
#pragma unroll
            for (int i = 0; i < 8; ++i)
                a_stage_f[i] =
                    *(const v4f*)(A + (size_t)(bm + a_r0 + 16 * i) * K + k0 + a_c4);
        } else {
            const bf16_t* A = (const bf16_t*)Aptr;
#pragma unroll
            for (int i = 0; i < 4; ++i)
                a_stage_h[i] =
                    *(const v8bf*)(A + (size_t)(bm + r0 + 32 * i) * K + k0 + c8);
        }
    };
    auto load_b = [&](int k0) {
#pragma unroll
        for (int i = 0; i < 4; ++i)
            b_stage[i] =
                *(const v8bf*)(Bt + (size_t)(bn + r0 + 32 * i) * K + k0 + c8);
    };
    auto store_tiles = [&](int buf) {
        if constexpr (A_F32) {
#pragma unroll
            for (int i = 0; i < 8; ++i) {
                v4bf b;
                b[0] = (bf16_t)a_stage_f[i][0];
                b[1] = (bf16_t)a_stage_f[i][1];
                b[2] = (bf16_t)a_stage_f[i][2];
                b[3] = (bf16_t)a_stage_f[i][3];
                *(v4bf*)(&sA[buf][(a_r0 + 16 * i) * LDT + a_c4]) = b;
            }
        } else {
#pragma unroll
            for (int i = 0; i < 4; ++i)
                *(v8bf*)(&sA[buf][(r0 + 32 * i) * LDT + c8]) = a_stage_h[i];
        }
#pragma unroll
        for (int i = 0; i < 4; ++i)
            *(v8bf*)(&sB[buf][(r0 + 32 * i) * LDT + c8]) = b_stage[i];
    };

    v8f acc[2][4];
#pragma unroll
    for (int i = 0; i < 2; ++i)
#pragma unroll
        for (int j = 0; j < 4; ++j)
#pragma unroll
            for (int r = 0; r < 8; ++r) acc[i][j][r] = 0.0f;

    // Prologue: stage tile 0 into buffer 0.
    load_a(0);
    load_b(0);
    store_tiles(0);
    __syncthreads();

    const int nt = K / BK;
    for (int kt = 0; kt < nt; ++kt) {
        const int buf = kt & 1;
        // Issue global loads for the NEXT tile before computing this one.
        if (kt + 1 < nt) {
            load_a((kt + 1) * BK);
            load_b((kt + 1) * BK);
        }

        // ---- 2 WMMA k-steps of K=32 each from LDS buffer `buf` ----
#pragma unroll
        for (int s = 0; s < 2; ++s) {
            v16bf afrag[2];
#pragma unroll
            for (int i = 0; i < 2; ++i) {
                // A 16x32 bf16 layout: lane half lh -> K base {0,8}; chunks
                // [b, b+8) and [b+16, b+24), row = M = lane%16.
                const bf16_t* p =
                    &sA[buf][(wm + i * 16 + lm) * LDT + s * 32 + lh * 8];
                v8bf c0 = *(const v8bf*)(p);
                v8bf c1 = *(const v8bf*)(p + 16);
                afrag[i] = __builtin_shufflevector(c0, c1, 0, 1, 2, 3, 4, 5, 6, 7,
                                                   8, 9, 10, 11, 12, 13, 14, 15);
            }
            v16bf bfrag[4];
#pragma unroll
            for (int j = 0; j < 4; ++j) {
                // B 32x16 bf16 layout: lane = column N, lane half lh -> K base
                // {0,8}; chunks [b, b+8) and [b+16, b+24).
                const bf16_t* p =
                    &sB[buf][(wn + j * 16 + lm) * LDT + s * 32 + lh * 8];
                v8bf c0 = *(const v8bf*)(p);
                v8bf c1 = *(const v8bf*)(p + 16);
                bfrag[j] = __builtin_shufflevector(c0, c1, 0, 1, 2, 3, 4, 5, 6, 7,
                                                   8, 9, 10, 11, 12, 13, 14, 15);
            }
#pragma unroll
            for (int i = 0; i < 2; ++i)
#pragma unroll
                for (int j = 0; j < 4; ++j)
                    acc[i][j] = __builtin_amdgcn_wmma_f32_16x16x32_bf16(
                        false, afrag[i], false, bfrag[j],
                        (short)0, acc[i][j], false, false);
        }

        // Drain staged registers into the alternate buffer, then one barrier.
        if (kt + 1 < nt) store_tiles(buf ^ 1);
        __syncthreads();
    }

    // ---- epilogue: C/D layout — VGPR r, lane: row = lh*8 + r, col = lane%16
#pragma unroll
    for (int i = 0; i < 2; ++i) {
#pragma unroll
        for (int j = 0; j < 4; ++j) {
#pragma unroll
            for (int r = 0; r < 8; ++r) {
                int row = bm + wm + i * 16 + lh * 8 + r;
                int col = bn + wn + j * 16 + lm;
                if constexpr (OUT_BF16) {
                    ((bf16_t*)Outp)[(size_t)row * N + col] = (bf16_t)acc[i][j][r];
                } else {
                    __builtin_nontemporal_store(
                        acc[i][j][r], (float*)Outp + (size_t)row * N + col);
                }
            }
        }
    }
}

// ---------------------------------------------------------------------------
// Launch: out = (x @ V) @ (U*sigma)^T, bf16 WMMA, t kept L2-resident.
// ---------------------------------------------------------------------------
extern "C" void kernel_launch(void* const* d_in, const int* in_sizes, int n_in,
                              void* d_out, int out_size, void* d_ws, size_t ws_size,
                              hipStream_t stream) {
    const float* x     = (const float*)d_in[0];   // [TOKENS][D_IN]
    const float* U     = (const float*)d_in[1];   // [D_OUT][RANK]
    const float* sigma = (const float*)d_in[2];   // [RANK]
    const float* V     = (const float*)d_in[3];   // [D_IN][RANK]
    float*       out   = (float*)d_out;           // [TOKENS][D_OUT]

    char* ws = (char*)d_ws;
    bf16_t* Us = (bf16_t*)(ws);                           // 4 MB  [D_OUT][RANK]
    bf16_t* Vt = (bf16_t*)(ws + (size_t)4 * 1024 * 1024); // 4 MB  [RANK][D_IN]
    bf16_t* t  = (bf16_t*)(ws + (size_t)8 * 1024 * 1024); // 16 MB [TOKENS][RANK]

    // Prep: Us = bf16(U * sigma);  Vt = bf16(V^T)
    k_scale_u<<<(D_OUT * RANK) / 256, 256, 0, stream>>>(U, sigma, Us);
    k_transpose_v<<<dim3(D_IN / 32, RANK / 32), 256, 0, stream>>>(V, Vt);

    // GEMM1: t[TOKENS][RANK] = x (f32->bf16) @ V    (B supplied as Vt=[N][K])
    k_gemm<true, true><<<dim3(RANK / 128, TOKENS / 128), 256, 0, stream>>>(
        (const void*)x, Vt, (void*)t, TOKENS, RANK, D_IN);

    // GEMM2: out[TOKENS][D_OUT] = t @ Us^T          (Us already [N][K])
    k_gemm<false, false><<<dim3(D_OUT / 128, TOKENS / 128), 256, 0, stream>>>(
        (const void*)t, Us, (void*)out, TOKENS, D_OUT, RANK);
}
